// CEIP_36155034697853
// MI455X (gfx1250) — matrix operand order repaired and checked by reading, not compiled, not run on hardware
//
#include <hip/hip_runtime.h>

typedef __attribute__((ext_vector_type(8)))  __bf16 v8bf;
typedef __attribute__((ext_vector_type(16))) __bf16 v16bf;
typedef __attribute__((ext_vector_type(8)))  float  v8f;

#define B_DIM 128
#define N_DIM 784
#define C_DIM 256
#define PITCH 264              // 256 bf16 + 16B pad per LDS row (stays 16B aligned)
#define INV_TEMP 10.0f         // 1 / 0.1
#define INV_CNT  (1.0f / (float)(B_DIM * N_DIM))

static __device__ __forceinline__ v16bf cat8(v8bf lo, v8bf hi) {
  return __builtin_shufflevector(lo, hi, 0,1,2,3,4,5,6,7,8,9,10,11,12,13,14,15);
}

// Async global->LDS 16B copy (CDNA5, tracked by ASYNCcnt; bypasses VGPRs).
// LDS aperture is selected by the high 32 bits of a flat address, so the low
// 32 bits of the generic shared pointer are the wave-relative LDS address.
static __device__ __forceinline__ void async_ld_lds_b128(void* lptr, const void* gptr) {
  unsigned lds_addr = (unsigned)(unsigned long long)lptr;
  asm volatile("global_load_async_to_lds_b128 %0, %1, off"
               :: "v"(lds_addr), "v"(gptr) : "memory");
}
static __device__ __forceinline__ void wait_async_zero() {
  asm volatile("s_wait_asynccnt 0" ::: "memory");
}

// A-fragment 16x32 bf16 (ISA 7.12.2): lane m = lane&15; lanes 0-15 hold
// K = [k0..k0+7, k0+16..k0+23], lanes 16-31 hold K = [k0+8..15, k0+24..31].
static __device__ __forceinline__ v16bf load_a_frag(const __bf16* A, int m0, int k0, int lane) {
  const __bf16* p = A + (m0 + (lane & 15)) * PITCH + k0 + ((lane >> 4) << 3);
  v8bf lo = *(const v8bf*)p;
  v8bf hi = *(const v8bf*)(p + 16);
  return cat8(lo, hi);
}

// B-fragment 32x16 bf16: lane col n = lane&15; lanes 0-15 hold K=[k0..k0+7, k0+16..k0+23],
// lanes 16-31 hold K=[k0+8..15, k0+24..31]. rowp points at element [n][k0] (K-contiguous).
static __device__ __forceinline__ v16bf load_b_frag(const __bf16* rowp, int lane) {
  const __bf16* p = rowp + ((lane >> 4) << 3);
  v8bf lo = *(const v8bf*)p;
  v8bf hi = *(const v8bf*)(p + 16);
  return cat8(lo, hi);
}

__global__ void ceip_zero_out(float* __restrict__ out) {
  if (threadIdx.x < 2) out[threadIdx.x] = 0.0f;
}

// One wave per (b,n) row of 256 floats: L2-normalize in fp32, emit bf16.
__global__ __launch_bounds__(256) void ceip_normalize(
    const float* __restrict__ zr, const float* __restrict__ ze,
    __bf16* __restrict__ nr, __bf16* __restrict__ ne) {
  const int lane = threadIdx.x & 31;
  const int wave = threadIdx.x >> 5;
  const size_t row = (size_t)blockIdx.x * 8 + wave;          // < B*N = 100352
  const float* src = (blockIdx.y == 0 ? zr : ze) + row * C_DIM + lane * 8;
  __bf16*      dst = (blockIdx.y == 0 ? nr : ne) + row * C_DIM + lane * 8;
  float4 a = ((const float4*)src)[0];
  float4 b = ((const float4*)src)[1];
  float s = a.x*a.x + a.y*a.y + a.z*a.z + a.w*a.w
          + b.x*b.x + b.y*b.y + b.z*b.z + b.w*b.w;
#pragma unroll
  for (int m = 16; m >= 1; m >>= 1) s += __shfl_xor(s, m, 32);
  const float inv = rsqrtf(s);
  v8bf o;
  o[0]=(__bf16)(a.x*inv); o[1]=(__bf16)(a.y*inv); o[2]=(__bf16)(a.z*inv); o[3]=(__bf16)(a.w*inv);
  o[4]=(__bf16)(b.x*inv); o[5]=(__bf16)(b.y*inv); o[6]=(__bf16)(b.z*inv); o[7]=(__bf16)(b.w*inv);
  *(v8bf*)dst = o;
}

// One workgroup per patch n: S = Zr_n(128x256) * Ze_n^T, fused logsumexp rows.
__global__ __launch_bounds__(256) void ceip_inter(
    const __bf16* __restrict__ nr, const __bf16* __restrict__ ne,
    float* __restrict__ out) {
  extern __shared__ __bf16 smem[];
  __bf16* Ar = smem;                    // 128 x PITCH
  __bf16* Be = smem + B_DIM * PITCH;    // 128 x PITCH
  const int n = blockIdx.x;
  const int tid = threadIdx.x;
  for (int idx = tid; idx < B_DIM * 32; idx += 256) {
    const int row = idx >> 5, ch = (idx & 31) << 3;          // 16B chunks
    const size_t g = ((size_t)row * N_DIM + n) * C_DIM + ch;
    async_ld_lds_b128(Ar + row * PITCH + ch, nr + g);
    async_ld_lds_b128(Be + row * PITCH + ch, ne + g);
  }
  wait_async_zero();
  __syncthreads();
  const int lane = tid & 31, wave = tid >> 5;
  const int m0 = wave << 4;                                   // 16 rows per wave
  v16bf afrag[8];
#pragma unroll
  for (int kt = 0; kt < 8; ++kt) afrag[kt] = load_a_frag(Ar, m0, kt * 32, lane);
  float rsum[8] = {0,0,0,0,0,0,0,0};
  float posv[8] = {0,0,0,0,0,0,0,0};
#pragma unroll
  for (int ct = 0; ct < 8; ++ct) {
    v8f acc = {};
    const __bf16* rowp = Be + ((ct << 4) + (lane & 15)) * PITCH;
#pragma unroll
    for (int kt = 0; kt < 8; ++kt) {
      v16bf bfrag = load_b_frag(rowp + kt * 32, lane);
      acc = __builtin_amdgcn_wmma_f32_16x16x32_bf16(false, afrag[kt], false, bfrag,
                                                    (short)0, acc, false, false);
    }
#pragma unroll
    for (int r = 0; r < 8; ++r) rsum[r] += __expf(acc[r] * INV_TEMP);
    if (ct == wave) {                                         // tile containing diagonal
      const int d = (lane & 15) - ((lane >> 4) << 3);
#pragma unroll
      for (int r = 0; r < 8; ++r)
        if (d == r) posv[r] = acc[r] * INV_TEMP;
    }
  }
#pragma unroll
  for (int r = 0; r < 8; ++r) {
#pragma unroll
    for (int m = 8; m >= 1; m >>= 1) {
      rsum[r] += __shfl_xor(rsum[r], m, 32);
      posv[r] += __shfl_xor(posv[r], m, 32);
    }
  }
  if ((lane & 15) == 0) {
    float local = 0.0f;
#pragma unroll
    for (int r = 0; r < 8; ++r) local += __logf(rsum[r]) - posv[r];
    atomicAdd(out, local * INV_CNT);
  }
}

// Grid (7 strips, 128 batches): S = Zr_b[strip](128x256) * Ze_b^T(256x784).
__global__ __launch_bounds__(256) void ceip_intra(
    const __bf16* __restrict__ nr, const __bf16* __restrict__ ne,
    float* __restrict__ out) {
  extern __shared__ __bf16 smem[];
  __bf16* Ar = smem;                                          // up to 128 x PITCH
  const int b = blockIdx.y;
  const int row0 = blockIdx.x * 128;                          // 0..768
  const int rows = min(128, N_DIM - row0);
  const int tid = threadIdx.x;
  const __bf16* zrb = nr + ((size_t)b * N_DIM + row0) * C_DIM;
  for (int idx = tid; idx < rows * 32; idx += 256) {
    const int row = idx >> 5, ch = (idx & 31) << 3;
    async_ld_lds_b128(Ar + row * PITCH + ch, zrb + (size_t)row * C_DIM + ch);
  }
  wait_async_zero();
  __syncthreads();
  const int lane = tid & 31, wave = tid >> 5;
  const int m0 = wave << 4;
  if (m0 >= rows) return;                                     // ragged last strip
  const __bf16* zeb = ne + (size_t)b * N_DIM * C_DIM;
  v16bf afrag[8];
#pragma unroll
  for (int kt = 0; kt < 8; ++kt) afrag[kt] = load_a_frag(Ar, m0, kt * 32, lane);
  float rsum[8] = {0,0,0,0,0,0,0,0};
  float posv[8] = {0,0,0,0,0,0,0,0};
  const int ctd = (row0 + m0) >> 4;                           // diagonal column tile
  for (int ct = 0; ct < N_DIM / 16; ++ct) {                   // 49 column tiles
    v8f acc = {};
    const __bf16* rowp = zeb + (size_t)((ct << 4) + (lane & 15)) * C_DIM;
    if (ct + 1 < N_DIM / 16)
      __builtin_prefetch(rowp + 16 * C_DIM, 0, 0);            // global_prefetch_b8
#pragma unroll
    for (int kt = 0; kt < 8; ++kt) {
      v16bf bfrag = load_b_frag(rowp + kt * 32, lane);
      acc = __builtin_amdgcn_wmma_f32_16x16x32_bf16(false, afrag[kt], false, bfrag,
                                                    (short)0, acc, false, false);
    }
#pragma unroll
    for (int r = 0; r < 8; ++r) rsum[r] += __expf(acc[r] * INV_TEMP);
    if (ct == ctd) {
      const int d = (lane & 15) - ((lane >> 4) << 3);
#pragma unroll
      for (int r = 0; r < 8; ++r)
        if (d == r) posv[r] = acc[r] * INV_TEMP;
    }
  }
#pragma unroll
  for (int r = 0; r < 8; ++r) {
#pragma unroll
    for (int m = 8; m >= 1; m >>= 1) {
      rsum[r] += __shfl_xor(rsum[r], m, 32);
      posv[r] += __shfl_xor(posv[r], m, 32);
    }
  }
  if ((lane & 15) == 0) {
    float local = 0.0f;
#pragma unroll
    for (int r = 0; r < 8; ++r) local += __logf(rsum[r]) - posv[r];
    atomicAdd(out + 1, local * INV_CNT);
  }
}

extern "C" void kernel_launch(void* const* d_in, const int* in_sizes, int n_in,
                              void* d_out, int out_size, void* d_ws, size_t ws_size,
                              hipStream_t stream) {
  (void)in_sizes; (void)n_in; (void)out_size; (void)ws_size;
  const float* zr = (const float*)d_in[0];
  const float* ze = (const float*)d_in[1];
  float* out = (float*)d_out;                                 // [inter_loss, intra_loss]
  __bf16* nr = (__bf16*)d_ws;                                 // normalized bf16 [B,N,C]
  __bf16* ne = nr + (size_t)B_DIM * N_DIM * C_DIM;

  const int SMEM_INTER = 2 * B_DIM * PITCH * (int)sizeof(__bf16);  // 135168 B
  const int SMEM_INTRA =     B_DIM * PITCH * (int)sizeof(__bf16);  //  67584 B
  (void)hipFuncSetAttribute((const void*)ceip_inter,
                            hipFuncAttributeMaxDynamicSharedMemorySize, SMEM_INTER);
  (void)hipFuncSetAttribute((const void*)ceip_intra,
                            hipFuncAttributeMaxDynamicSharedMemorySize, SMEM_INTRA);

  ceip_zero_out<<<1, 32, 0, stream>>>(out);
  ceip_normalize<<<dim3((B_DIM * N_DIM) / 8, 2), 256, 0, stream>>>(zr, ze, nr, ne);
  ceip_inter<<<N_DIM, 256, SMEM_INTER, stream>>>(nr, ne, out);
  ceip_intra<<<dim3(N_DIM / 112, B_DIM), 256, SMEM_INTRA, stream>>>(nr, ne, out);
}
